// GATLayer_36928128811112
// MI455X (gfx1250) — matrix operand order repaired and checked by reading, not compiled, not run on hardware
//
#include <hip/hip_runtime.h>
#include <hip/hip_bf16.h>

// ---- problem constants ----
#define NN   4096
#define HH   8
#define FIN  128
#define FOUT 64

typedef __attribute__((ext_vector_type(16))) _Float16 v16h;
typedef __attribute__((ext_vector_type(8)))  _Float16 v8h;
typedef __attribute__((ext_vector_type(8)))  float    v8f;
typedef __attribute__((ext_vector_type(4)))  float    v4f;

union U16 { v16h v; _Float16 e[16]; };
union V8F { v4f v[2]; float f[8]; };

__device__ __forceinline__ v8f zero8() {
    v8f z = {0.f,0.f,0.f,0.f,0.f,0.f,0.f,0.f};
    return z;
}

// ---------------------------------------------------------------------------
// f32 -> f16 conversions
// ---------------------------------------------------------------------------
__global__ void k_cvt_x(const float* __restrict__ x, _Float16* __restrict__ xh) {
    int i = blockIdx.x * blockDim.x + threadIdx.x;
    if (i < NN * FIN) xh[i] = (_Float16)x[i];
}

__global__ void k_cvt_skw(const float* __restrict__ w, _Float16* __restrict__ wh) {
    int i = blockIdx.x * blockDim.x + threadIdx.x;
    if (i < (HH * FOUT) * FIN) wh[i] = (_Float16)w[i];
}

// proj [h][f][o] -> projT [h][o][f] (f16) so B fragments load contiguously in f
__global__ void k_cvt_projT(const float* __restrict__ proj, _Float16* __restrict__ pT) {
    int i = blockIdx.x * blockDim.x + threadIdx.x;
    if (i >= HH * FIN * FOUT) return;
    int h = i >> 13;          // /(128*64)
    int r = i & 8191;
    int f = r >> 6;
    int o = r & 63;
    pT[(h * FOUT + o) * FIN + f] = (_Float16)proj[i];
}

// ---------------------------------------------------------------------------
// Projection: P_h = X @ proj_h via WMMA; emits Pt (f16, [h][o][n]) and
// s_src/s_tgt (f32, [h][n]) reduced from the accumulators in-register.
// One wave per (head, 16-row block).
// ---------------------------------------------------------------------------
__global__ __launch_bounds__(32) void k_proj(
    const _Float16* __restrict__ xh, const _Float16* __restrict__ projT,
    const float* __restrict__ a_src, const float* __restrict__ a_tgt,
    _Float16* __restrict__ Pt, float* __restrict__ s_src, float* __restrict__ s_tgt)
{
    const int h = blockIdx.y;
    const int rowbase = blockIdx.x * 16;
    const int lane = threadIdx.x;
    const int lm   = lane & 15;
    const int hi8  = (lane < 16) ? 0 : 8;

    v8f acc[4] = {zero8(), zero8(), zero8(), zero8()};

    #pragma unroll
    for (int ks = 0; ks < FIN; ks += 32) {
        // A fragment: 16x32 f16. lane-half k split: 0-15 vs 16-31
        U16 a;
        const _Float16* ap = xh + (rowbase + lm) * FIN + ks + ((lane < 16) ? 0 : 16);
        *(v8h*)&a.e[0] = *(const v8h*)(ap);
        *(v8h*)&a.e[8] = *(const v8h*)(ap + 8);
        #pragma unroll
        for (int t = 0; t < 4; ++t) {
            // B fragment: 32x16, lane-half K = 0..15 / 16..31, contiguous in f
            const _Float16* bp = projT + (h * FOUT + t * 16 + lm) * FIN + ks + ((lane < 16) ? 0 : 16);
            v16h b = *(const v16h*)bp;
            acc[t] = __builtin_amdgcn_wmma_f32_16x16x32_f16(
                false, a.v, false, b, (short)0, acc[t], false, false);
        }
    }

    // s_src / s_tgt: dot accumulator rows with attention vectors, reduce over
    // the 16 lanes of each half (columns) via shfl_xor.
    float asv[4], atv[4];
    #pragma unroll
    for (int t = 0; t < 4; ++t) {
        asv[t] = a_src[h * FOUT + t * 16 + lm];
        atv[t] = a_tgt[h * FOUT + t * 16 + lm];
    }
    float ps[8], pt[8];
    #pragma unroll
    for (int i = 0; i < 8; ++i) {
        float s = 0.f, u = 0.f;
        #pragma unroll
        for (int t = 0; t < 4; ++t) { s += acc[t][i] * asv[t]; u += acc[t][i] * atv[t]; }
        ps[i] = s; pt[i] = u;
    }
    #pragma unroll
    for (int mask = 1; mask < 16; mask <<= 1)
        #pragma unroll
        for (int i = 0; i < 8; ++i) {
            ps[i] += __shfl_xor(ps[i], mask, 32);
            pt[i] += __shfl_xor(pt[i], mask, 32);
        }
    if (lm == 0) {
        #pragma unroll
        for (int i = 0; i < 8; ++i) {
            s_src[h * NN + rowbase + hi8 + i] = ps[i];
            s_tgt[h * NN + rowbase + hi8 + i] = pt[i];
        }
    }

    // store Pt transposed as f16: Pt[h][o][n]
    #pragma unroll
    for (int t = 0; t < 4; ++t)
        #pragma unroll
        for (int i = 0; i < 8; ++i)
            Pt[(h * FOUT + t * 16 + lm) * NN + rowbase + hi8 + i] = (_Float16)acc[t][i];
}

// ---------------------------------------------------------------------------
// Skip projection: sk[n][j] = x[n,:] . skip_w[j,:]  (j = h*64+o), via WMMA.
// skip_w is [512][128] row-major -> already the transposed B we need.
// ---------------------------------------------------------------------------
__global__ __launch_bounds__(32) void k_skip(
    const _Float16* __restrict__ xh, const _Float16* __restrict__ skw,
    float* __restrict__ sk)
{
    const int rowbase = blockIdx.x * 16;
    const int jbase   = blockIdx.y * 64;
    const int lane = threadIdx.x;
    const int lm   = lane & 15;
    const int hi8  = (lane < 16) ? 0 : 8;

    v8f acc[4] = {zero8(), zero8(), zero8(), zero8()};

    #pragma unroll
    for (int ks = 0; ks < FIN; ks += 32) {
        U16 a;
        const _Float16* ap = xh + (rowbase + lm) * FIN + ks + ((lane < 16) ? 0 : 16);
        *(v8h*)&a.e[0] = *(const v8h*)(ap);
        *(v8h*)&a.e[8] = *(const v8h*)(ap + 8);
        #pragma unroll
        for (int t = 0; t < 4; ++t) {
            const _Float16* bp = skw + (jbase + t * 16 + lm) * FIN + ks + ((lane < 16) ? 0 : 16);
            v16h b = *(const v16h*)bp;
            acc[t] = __builtin_amdgcn_wmma_f32_16x16x32_f16(
                false, a.v, false, b, (short)0, acc[t], false, false);
        }
    }
    #pragma unroll
    for (int t = 0; t < 4; ++t)
        #pragma unroll
        for (int i = 0; i < 8; ++i)
            sk[(rowbase + hi8 + i) * (HH * FOUT) + jbase + t * 16 + lm] = acc[t][i];
}

// ---------------------------------------------------------------------------
// Fused masked-softmax attention + aggregation (flash-style, no [H,N,N]).
// 256 threads = 8 waves; wave w handles head w; topology tile shared via LDS
// with a double buffer (one barrier per k-tile, staging overlaps compute).
// Scores are bounded so no running-max is needed: exp(-1e9) == 0 exactly.
// ---------------------------------------------------------------------------
__global__ __launch_bounds__(256) void k_attn(
    const float* __restrict__ topo, const float* __restrict__ s_src,
    const float* __restrict__ s_tgt, const _Float16* __restrict__ Pt,
    float* __restrict__ agg)
{
    __shared__ float lds_topo[2][16 * 32];   // double-buffered mask tile
    __shared__ float lds_stgt[2][HH * 32];   // double-buffered s_tgt slice

    const int tid  = threadIdx.x;
    const int w    = tid >> 5;            // head
    const int lane = tid & 31;
    const int lm   = lane & 15;
    const int hi8  = (lane < 16) ? 0 : 8;
    const int koff = (lane < 16) ? 0 : 16; // A-fragment lane-half K offset
    const int rowbase = blockIdx.x * 16;

    const int sr  = tid >> 5;             // staging row / head index
    const int sk_ = tid & 31;             // staging column index

    const float s_row = s_src[w * NN + rowbase + lm];

    v8f acc[4] = {zero8(), zero8(), zero8(), zero8()};
    float rsum = 0.f;

    // prologue: stage tile 0 into buffer 0
    lds_topo[0][tid]       = topo[(rowbase +     sr) * NN + sk_];
    lds_topo[0][tid + 256] = topo[(rowbase + 8 + sr) * NN + sk_];
    lds_stgt[0][tid]       = s_tgt[sr * NN + sk_];

    for (int it = 0; it < NN / 32; ++it) {
        const int m = it * 32;
        const int p = it & 1;
        __syncthreads();   // buf[p] staged; all reads of buf[p^1] (iter it-1) done

        // stage next tile into the other buffer (block-uniform condition)
        if (m + 32 < NN) {
            lds_topo[p ^ 1][tid]       = topo[(rowbase +     sr) * NN + m + 32 + sk_];
            lds_topo[p ^ 1][tid + 256] = topo[(rowbase + 8 + sr) * NN + m + 32 + sk_];
            lds_stgt[p ^ 1][tid]       = s_tgt[sr * NN + m + 32 + sk_];
            if (m + 64 < NN)
                __builtin_prefetch(&topo[(rowbase + sr) * NN + m + 64 + sk_], 0, 1);
        }

        // vectorized LDS reads: two contiguous 8-float runs per lane per array
        V8F tlo, thi, slo, shi;
        tlo.v[0] = *(const v4f*)&lds_topo[p][lm * 32 + koff];
        tlo.v[1] = *(const v4f*)&lds_topo[p][lm * 32 + koff + 4];
        thi.v[0] = *(const v4f*)&lds_topo[p][lm * 32 + koff + 8];
        thi.v[1] = *(const v4f*)&lds_topo[p][lm * 32 + koff + 12];
        slo.v[0] = *(const v4f*)&lds_stgt[p][w * 32 + koff];
        slo.v[1] = *(const v4f*)&lds_stgt[p][w * 32 + koff + 4];
        shi.v[0] = *(const v4f*)&lds_stgt[p][w * 32 + koff + 8];
        shi.v[1] = *(const v4f*)&lds_stgt[p][w * 32 + koff + 12];

        // Build 16x32 probability A-fragment in registers.
        // element j (0..7)  <-> k = koff + j        (tlo/slo)
        // element j (8..15) <-> k = koff + 8 + j-8  (thi/shi)
        U16 a;
        #pragma unroll
        for (int j = 0; j < 8; ++j) {
            float e0 = s_row + slo.f[j];
            e0 = fmaxf(e0, 0.2f * e0);       // leaky_relu(0.2)
            e0 += tlo.f[j];                  // 0 or -1e9 mask
            float p0 = __expf(e0);           // masked -> exactly 0
            rsum += p0;
            a.e[j] = (_Float16)p0;

            float e1 = s_row + shi.f[j];
            e1 = fmaxf(e1, 0.2f * e1);
            e1 += thi.f[j];
            float p1 = __expf(e1);
            rsum += p1;
            a.e[8 + j] = (_Float16)p1;
        }

        #pragma unroll
        for (int t = 0; t < 4; ++t) {
            const _Float16* bp = Pt + (w * FOUT + t * 16 + lm) * NN + m + ((lane < 16) ? 0 : 16);
            v16h b = *(const v16h*)bp;
            acc[t] = __builtin_amdgcn_wmma_f32_16x16x32_f16(
                false, a.v, false, b, (short)0, acc[t], false, false);
        }
    }

    // complete per-row softmax denominators (row r lives in lanes r and r+16)
    rsum += __shfl_xor(rsum, 16, 32);
    float inv[8];
    #pragma unroll
    for (int i = 0; i < 8; ++i) {
        float rs = __shfl(rsum, hi8 + i, 32);
        inv[i] = 1.0f / rs;
    }

    #pragma unroll
    for (int t = 0; t < 4; ++t)
        #pragma unroll
        for (int i = 0; i < 8; ++i)
            agg[(w * NN + rowbase + hi8 + i) * FOUT + t * 16 + lm] = acc[t][i] * inv[i];
}

// ---------------------------------------------------------------------------
// out[n,o] = lrelu( mean_h( agg[h,n,o] + sk[n, h*64+o] ) )
// ---------------------------------------------------------------------------
__global__ void k_final(const float* __restrict__ agg, const float* __restrict__ sk,
                        float* __restrict__ out)
{
    int i = blockIdx.x * blockDim.x + threadIdx.x;
    if (i >= NN * FOUT) return;
    int n = i >> 6, o = i & 63;
    float s = 0.f;
    #pragma unroll
    for (int h = 0; h < HH; ++h)
        s += agg[(h * NN + n) * FOUT + o] + sk[n * (HH * FOUT) + h * FOUT + o];
    float v = s * (1.0f / HH);
    out[i] = fmaxf(v, 0.2f * v);
}

// ---------------------------------------------------------------------------
extern "C" void kernel_launch(void* const* d_in, const int* in_sizes, int n_in,
                              void* d_out, int out_size, void* d_ws, size_t ws_size,
                              hipStream_t stream) {
    (void)in_sizes; (void)n_in; (void)out_size; (void)ws_size;
    const float* x     = (const float*)d_in[0];
    const float* topo  = (const float*)d_in[1];
    const float* proj  = (const float*)d_in[2];
    const float* a_src = (const float*)d_in[3];
    const float* a_tgt = (const float*)d_in[4];
    const float* skw32 = (const float*)d_in[5];
    float* out = (float*)d_out;

    // workspace layout (all offsets 4KB-aligned); total ~21.5 MB
    char* ws = (char*)d_ws;
    _Float16* xh    = (_Float16*)(ws);                         // 1 MB
    _Float16* projT = (_Float16*)(ws + 1048576);               // 128 KB
    _Float16* skw   = (_Float16*)(ws + 1048576 + 131072);      // 128 KB
    _Float16* Pt    = (_Float16*)(ws + 1310720);               // 4 MB
    float*    ssrc  = (float*)   (ws + 1310720 + 4194304);     // 128 KB
    float*    stgt  = (float*)   (ws + 5505024 + 131072);      // 128 KB
    float*    agg   = (float*)   (ws + 5505024 + 262144);      // 8 MB
    float*    sk    = (float*)   (ws + 5767168 + 8388608);     // 8 MB

    // conversions
    k_cvt_x    <<<(NN * FIN + 255) / 256, 256, 0, stream>>>(x, xh);
    k_cvt_projT<<<(HH * FIN * FOUT + 255) / 256, 256, 0, stream>>>(proj, projT);
    k_cvt_skw  <<<(HH * FOUT * FIN + 255) / 256, 256, 0, stream>>>(skw32, skw);

    // projection (WMMA) + attention score vectors
    k_proj<<<dim3(NN / 16, HH), 32, 0, stream>>>(xh, projT, a_src, a_tgt, Pt, ssrc, stgt);

    // skip projection (WMMA)
    k_skip<<<dim3(NN / 16, HH), 32, 0, stream>>>(xh, skw, sk);

    // fused masked softmax + aggregation (WMMA)
    k_attn<<<NN / 16, 256, 0, stream>>>(topo, ssrc, stgt, Pt, agg);

    // head mean + skip + leaky relu
    k_final<<<(NN * FOUT + 255) / 256, 256, 0, stream>>>(agg, sk, out);
}
